// LSTMTagger_64879775973938
// MI455X (gfx1250) — matrix operand order repaired and checked by reading, not compiled, not run on hardware
//
#include <hip/hip_runtime.h>
#include <hip/hip_bf16.h>

typedef __attribute__((ext_vector_type(16))) _Float16 v16h;
typedef __attribute__((ext_vector_type(8)))  float    v8f;

#define WMMA_F16(a, b, c) \
  __builtin_amdgcn_wmma_f32_16x16x32_f16(false, (a), false, (b), (short)0, (c), false, false)

__device__ __forceinline__ v8f vzero8() {
  v8f v;
#pragma unroll
  for (int i = 0; i < 8; ++i) v[i] = 0.f;
  return v;
}

__device__ __forceinline__ float sigf(float x) { return 1.f / (1.f + __expf(-x)); }

// ---- WMMA fragment loaders (per CDNA5 ISA 7.12.2 layouts) -------------------
// A: 16x32 f16 tile of A[m][k], row-major, leading dim ld.
__device__ __forceinline__ v16h load_a_frag(const _Float16* __restrict__ base,
                                            int row0, int ld, int k0) {
  int lane = threadIdx.x & 31;
  int m  = row0 + (lane & 15);
  int kb = k0 + ((lane >> 4) << 3);
  const _Float16* p = base + (size_t)m * ld + kb;
  v16h a;
#pragma unroll
  for (int i = 0; i < 8; ++i) a[i] = p[i];
#pragma unroll
  for (int i = 0; i < 8; ++i) a[8 + i] = p[16 + i];
  return a;
}

// B: swizzled fragment-major tiles. Tile (ktile, ntile) of a (Kpad x Ntot)
// K-major matrix is stored as 32 lanes x 16 halves contiguous (1 KB), so each
// lane reads 32 contiguous bytes -> 2x global_load_b128.
__device__ __forceinline__ v16h load_b_frag_sw(const _Float16* __restrict__ base,
                                               int ktile, int ntiles, int ntile) {
  int lane = threadIdx.x & 31;
  const _Float16* p = base + (((size_t)ktile * ntiles + ntile) << 9) + (lane << 4);
  v16h b;
#pragma unroll
  for (int i = 0; i < 16; ++i) b[i] = p[i];
  return b;
}

// ---- weight conversion: f32 (Mout x Kin) row-major -> f16 swizzled tiles ----
// dst has Ntot columns total; this call fills columns [moff, moff+Mpad).
__global__ void wconvert_swz_kernel(const float* __restrict__ src, _Float16* __restrict__ dst,
                                    int Mout, int Kin, int Kpad, int Mpad, int Ntot, int moff) {
  int id = blockIdx.x * blockDim.x + threadIdx.x;
  int total = Kpad * Mpad;
  if (id >= total) return;
  int k = id / Mpad, m = id - k * Mpad;
  float v = (k < Kin && m < Mout) ? src[(size_t)m * Kin + k] : 0.f;
  int col = moff + m;
  int ntiles = Ntot >> 4;
  int kt = k >> 5, nt = col >> 4;
  int kr = k & 31;
  int lane = (((kr >> 3) & 1) << 4) + (col & 15);
  int j = (kr & 7) + ((kr >> 4) << 3);
  dst[(((size_t)kt * ntiles + nt) << 9) + (lane << 4) + j] = (_Float16)v;
}

// ---- char embedding gather: x (t, n, 64) f16 --------------------------------
__global__ void charembed_kernel(const int* __restrict__ cmap, const float* __restrict__ ceW,
                                 _Float16* __restrict__ x) {
  int id = blockIdx.x * blockDim.x + threadIdx.x;  // 512*64*64
  if (id >= 512 * 64 * 64) return;
  int k = id & 63, n = (id >> 6) & 63, t = id >> 12;
  x[id] = (_Float16)ceW[(size_t)cmap[n * 512 + t] * 64 + k];
}

// ---- persistent char LSTM: 512 steps, 16 batch rows per block ---------------
// grid (4 row-blocks, 2 dirs), 512 threads (16 waves, one j-tile per wave).
__global__ __launch_bounds__(512) void char_lstm_kernel(
    const _Float16* __restrict__ xF, const _Float16* __restrict__ xB,
    const _Float16* __restrict__ WihF, const _Float16* __restrict__ WhhF,
    const float* __restrict__ bF, const float* __restrict__ mF, _Float16* __restrict__ hsF,
    const _Float16* __restrict__ WihB, const _Float16* __restrict__ WhhB,
    const float* __restrict__ bB, const float* __restrict__ mB, _Float16* __restrict__ hsB) {
  int dir = blockIdx.y;
  const _Float16* x    = dir ? xB : xF;
  const _Float16* Wih  = dir ? WihB : WihF;
  const _Float16* Whh  = dir ? WhhB : WhhF;
  const float*    bias = dir ? bB : bF;
  const float*    mask = dir ? mB : mF;
  _Float16*       hs   = dir ? hsB : hsF;
  int rowBase = blockIdx.x * 16;

  __shared__ _Float16 hlds[16 * 256];  // 8 KB
  int tid = threadIdx.x, wave = tid >> 5, lane = tid & 31;
  int jt = wave;  // 16 j-tiles
  for (int i = tid; i < 16 * 256; i += 512) hlds[i] = (_Float16)0.f;
  v8f cst = vzero8();
  float bvals[4];
#pragma unroll
  for (int g = 0; g < 4; ++g) bvals[g] = bias[g * 256 + jt * 16 + (lane & 15)];
  // Wih fragments are tiny (2 ktiles x 4 gates): keep resident in registers
  v16h wihf[2][4];
#pragma unroll
  for (int kt = 0; kt < 2; ++kt)
#pragma unroll
    for (int g = 0; g < 4; ++g) wihf[kt][g] = load_b_frag_sw(Wih, kt, 64, g * 16 + jt);
  __syncthreads();
  int jcol = jt * 16 + (lane & 15);

  for (int t = 0; t < 512; ++t) {
    const _Float16* xt = x + (size_t)t * 64 * 64 + (size_t)rowBase * 64;
    v8f z[4];
#pragma unroll
    for (int g = 0; g < 4; ++g) z[g] = vzero8();
    // x @ Wih^T : K = 64 (weights in registers)
#pragma unroll
    for (int kt = 0; kt < 2; ++kt) {
      v16h a = load_a_frag(xt, 0, 64, kt * 32);
#pragma unroll
      for (int g = 0; g < 4; ++g) z[g] = WMMA_F16(a, wihf[kt][g], z[g]);
    }
    // h @ Whh^T : K = 256 (h from LDS, weights streamed from L2)
    for (int kt = 0; kt < 8; ++kt) {
      v16h a = load_a_frag(hlds, 0, 256, kt * 32);
#pragma unroll
      for (int g = 0; g < 4; ++g) {
        v16h bb = load_b_frag_sw(Whh, kt, 64, g * 16 + jt);
        z[g] = WMMA_F16(a, bb, z[g]);
      }
    }
    __syncthreads();  // all reads of old h complete
#pragma unroll
    for (int e = 0; e < 8; ++e) {
      int nl = ((lane >> 4) << 3) + e;
      int n = rowBase + nl;
      float zi = z[0][e] + bvals[0];
      float zf = z[1][e] + bvals[1];
      float zg = z[2][e] + bvals[2];
      float zo = z[3][e] + bvals[3];
      float c = sigf(zf) * cst[e] + sigf(zi) * tanhf(zg);
      float h = sigf(zo) * tanhf(c);
      cst[e] = c;
      hlds[nl * 256 + jcol] = (_Float16)h;
      hs[(size_t)t * 64 * 256 + (size_t)n * 256 + jcol] =
          (_Float16)(h * mask[n * 512 + t]);
    }
    __syncthreads();  // new h visible before next step
  }
}

// ---- stable compaction: one block per (b, dir), T=128 kept rows -------------
__global__ __launch_bounds__(512) void compact_kernel(const _Float16* __restrict__ hs_f,
                                                      const _Float16* __restrict__ hs_b,
                                                      _Float16* __restrict__ sub) {
  __shared__ int scanbuf[512];
  int b = blockIdx.x, dir = blockIdx.y, t = threadIdx.x;
  const _Float16* hs = dir ? hs_b : hs_f;
  const _Float16* row = hs + (size_t)t * 64 * 256 + (size_t)b * 256;
  float s = 0.f;
  for (int j = 0; j < 256; ++j) s += (float)row[j];
  int flag = (s != 0.f) ? 1 : 0;
  scanbuf[t] = flag;
  __syncthreads();
  for (int off = 1; off < 512; off <<= 1) {
    int add = (t >= off) ? scanbuf[t - off] : 0;
    __syncthreads();
    scanbuf[t] += add;
    __syncthreads();
  }
  int incl = scanbuf[t], total = scanbuf[511];
  int dest = flag ? (incl - 1) : (total + (t - incl));
  if (dest < 128) {
    _Float16* drow = sub + ((size_t)(b * 128 + dest)) * 512 + dir * 256;
    for (int j = 0; j < 256; ++j) drow[j] = row[j];
  }
}

// ---- highway GEMM: (8192 x 512) x (512 x 1024) -> f32 -----------------------
__global__ __launch_bounds__(512) void hwgemm_kernel(const _Float16* __restrict__ A,
                                                     const _Float16* __restrict__ B,
                                                     float* __restrict__ C) {
  int gid = blockIdx.x * 16 + (threadIdx.x >> 5);
  int lane = threadIdx.x & 31;
  int rt = gid >> 6, ct = gid & 63;
  v8f acc = vzero8();
  for (int kt = 0; kt < 16; ++kt) {
    v16h a = load_a_frag(A, rt * 16, 512, kt * 32);
    v16h bb = load_b_frag_sw(B, kt, 64, ct);
    acc = WMMA_F16(a, bb, acc);
  }
  int col = ct * 16 + (lane & 15);
  int rbase = rt * 16 + ((lane >> 4) << 3);
#pragma unroll
  for (int e = 0; e < 8; ++e) C[(size_t)(rbase + e) * 1024 + col] = acc[e];
}

// ---- word embedding + pad of w buffer (r, 640) ------------------------------
__global__ void wordembed_kernel(const int* __restrict__ wmap, const float* __restrict__ weW,
                                 _Float16* __restrict__ wbuf) {
  int id = blockIdx.x * blockDim.x + threadIdx.x;  // 8192 * 128
  if (id >= 8192 * 128) return;
  int r = id >> 7, k = id & 127;
  if (k < 100)
    wbuf[(size_t)r * 640 + k] = (_Float16)weW[(size_t)wmap[r] * 100 + k];
  else
    wbuf[(size_t)r * 640 + 512 + k] = (_Float16)0.f;  // pads 612..639
}

// ---- highway elementwise: writes sub' into w buffer cols 100..611 -----------
__global__ void highway_kernel(const float* __restrict__ TG, const _Float16* __restrict__ sub,
                               const float* __restrict__ bt, const float* __restrict__ bg,
                               _Float16* __restrict__ wbuf) {
  int id = blockIdx.x * blockDim.x + threadIdx.x;  // 8192*512
  if (id >= 8192 * 512) return;
  int r = id >> 9, j = id & 511;
  float tr = TG[(size_t)r * 1024 + j] + bt[j];
  tr = fmaxf(tr, 0.f);
  float g = sigf(TG[(size_t)r * 1024 + 512 + j] + bg[j]);
  float s = (float)sub[(size_t)r * 512 + j];
  wbuf[(size_t)r * 640 + 100 + j] = (_Float16)(g * tr + (1.f - g) * s);
}

// ---- persistent word LSTM: 64 steps over b, 16 "batch" rows per block -------
// grid (8 row-blocks, 2 dirs), 512 threads (16 waves, one j-tile per wave).
__global__ __launch_bounds__(512) void word_lstm_kernel(
    const _Float16* __restrict__ w,  // (64, 128, 640)
    const _Float16* __restrict__ WihF, const _Float16* __restrict__ WhhF,
    const float* __restrict__ bF,
    const _Float16* __restrict__ WihB, const _Float16* __restrict__ WhhB,
    const float* __restrict__ bB,
    _Float16* __restrict__ wh) {  // (64, 128, 512)
  int dir = blockIdx.y;
  const _Float16* Wih  = dir ? WihB : WihF;
  const _Float16* Whh  = dir ? WhhB : WhhF;
  const float*    bias = dir ? bB : bF;
  int colOff = dir * 256;
  int rowBase = blockIdx.x * 16;

  __shared__ _Float16 hlds[16 * 256];  // 8 KB
  int tid = threadIdx.x, wave = tid >> 5, lane = tid & 31;
  int jt = wave;
  for (int i = tid; i < 16 * 256; i += 512) hlds[i] = (_Float16)0.f;
  v8f cst = vzero8();
  float bvals[4];
#pragma unroll
  for (int g = 0; g < 4; ++g) bvals[g] = bias[g * 256 + jt * 16 + (lane & 15)];
  __syncthreads();
  int jcol = jt * 16 + (lane & 15);

  for (int s = 0; s < 64; ++s) {
    int b = dir ? (63 - s) : s;
    const _Float16* xb = w + (size_t)b * 128 * 640 + (size_t)rowBase * 640;
    v8f z[4];
#pragma unroll
    for (int g = 0; g < 4; ++g) z[g] = vzero8();
    // x @ Wih^T : K = 640 (612 padded)
    for (int kt = 0; kt < 20; ++kt) {
      v16h a = load_a_frag(xb, 0, 640, kt * 32);
#pragma unroll
      for (int g = 0; g < 4; ++g) {
        v16h bb = load_b_frag_sw(Wih, kt, 64, g * 16 + jt);
        z[g] = WMMA_F16(a, bb, z[g]);
      }
    }
    // h @ Whh^T : K = 256 (h from LDS)
    for (int kt = 0; kt < 8; ++kt) {
      v16h a = load_a_frag(hlds, 0, 256, kt * 32);
#pragma unroll
      for (int g = 0; g < 4; ++g) {
        v16h bb = load_b_frag_sw(Whh, kt, 64, g * 16 + jt);
        z[g] = WMMA_F16(a, bb, z[g]);
      }
    }
    __syncthreads();  // all reads of old h complete
#pragma unroll
    for (int e = 0; e < 8; ++e) {
      int nl = ((lane >> 4) << 3) + e;
      int n = rowBase + nl;
      float zi = z[0][e] + bvals[0];
      float zf = z[1][e] + bvals[1];
      float zg = z[2][e] + bvals[2];
      float zo = z[3][e] + bvals[3];
      float c = sigf(zf) * cst[e] + sigf(zi) * tanhf(zg);
      float h = sigf(zo) * tanhf(c);
      cst[e] = c;
      hlds[nl * 256 + jcol] = (_Float16)h;
      wh[((size_t)b * 128 + n) * 512 + colOff + jcol] = (_Float16)h;
    }
    __syncthreads();
  }
}

// ---- emission GEMM (tags padded to 64) fused with CRF broadcast -------------
__global__ __launch_bounds__(256) void emission_kernel(const _Float16* __restrict__ wh,
                                                       const _Float16* __restrict__ emW,
                                                       const float* __restrict__ emb,
                                                       const float* __restrict__ trans,
                                                       float* __restrict__ out) {
  int rt = blockIdx.x * 8 + (threadIdx.x >> 5);  // 512 row tiles
  int lane = threadIdx.x & 31;
  v8f acc[4];
#pragma unroll
  for (int c = 0; c < 4; ++c) acc[c] = vzero8();
  for (int kt = 0; kt < 16; ++kt) {
    v16h a = load_a_frag(wh, rt * 16, 512, kt * 32);
#pragma unroll
    for (int c = 0; c < 4; ++c) {
      v16h bb = load_b_frag_sw(emW, kt, 4, c);
      acc[c] = WMMA_F16(a, bb, acc[c]);
    }
  }
#pragma unroll
  for (int ct = 0; ct < 4; ++ct) {
    int col = ct * 16 + (lane & 15);
    if (col < 50) {
      float bias = emb[col];
#pragma unroll
      for (int e = 0; e < 8; ++e) {
        int row = rt * 16 + ((lane >> 4) << 3) + e;  // row = b*128 + t
        float ev = acc[ct][e] + bias;
        float* dst = out + (size_t)row * 50 * 50 + col;
        for (int i = 0; i < 50; ++i) dst[(size_t)i * 50] = ev + trans[i * 50 + col];
      }
    }
  }
}

// ---- tail: tmaps, lengths, tmap_correct appended as floats ------------------
__global__ void tail_kernel(const int* __restrict__ tmaps, const int* __restrict__ lengths,
                            const int* __restrict__ tmc, float* __restrict__ out) {
  int id = blockIdx.x * blockDim.x + threadIdx.x;
  const size_t OFF = 20480000;  // 64*128*50*50
  if (id < 8192) out[OFF + id] = (float)tmaps[id];
  else if (id < 8256) out[OFF + id] = (float)lengths[id - 8192];
  else if (id < 16448) out[OFF + id] = (float)tmc[id - 8256];
}

extern "C" void kernel_launch(void* const* d_in, const int* in_sizes, int n_in,
                              void* d_out, int out_size, void* d_ws, size_t ws_size,
                              hipStream_t stream) {
  const int*   wmap      = (const int*)d_in[0];
  const int*   cmaps_f   = (const int*)d_in[1];
  const int*   cmaps_b   = (const int*)d_in[2];
  const float* cmakers_f = (const float*)d_in[3];
  const float* cmakers_b = (const float*)d_in[4];
  const int*   tmaps     = (const int*)d_in[5];
  const int*   tmap_corr = (const int*)d_in[6];
  const int*   lengths   = (const int*)d_in[7];
  const float* ce_W      = (const float*)d_in[8];
  const float* fc_Wih = (const float*)d_in[9],  *fc_Whh = (const float*)d_in[10], *fc_b = (const float*)d_in[11];
  const float* bc_Wih = (const float*)d_in[12], *bc_Whh = (const float*)d_in[13], *bc_b = (const float*)d_in[14];
  const float* we_W   = (const float*)d_in[15];
  const float* wf_Wih = (const float*)d_in[16], *wf_Whh = (const float*)d_in[17], *wf_b = (const float*)d_in[18];
  const float* wb_Wih = (const float*)d_in[19], *wb_Whh = (const float*)d_in[20], *wb_b = (const float*)d_in[21];
  const float* hw_Wt = (const float*)d_in[22], *hw_bt = (const float*)d_in[23];
  const float* hw_Wg = (const float*)d_in[24], *hw_bg = (const float*)d_in[25];
  const float* em_W = (const float*)d_in[26], *em_b = (const float*)d_in[27];
  const float* trans = (const float*)d_in[28];
  float* out = (float*)d_out;

  // workspace layout
  char* base = (char*)d_ws;
  size_t off = 0;
  auto alloc = [&](size_t bytes) -> void* {
    void* p = base + off;
    off = (off + bytes + 255) & ~(size_t)255;
    return p;
  };
  _Float16* xc_f   = (_Float16*)alloc((size_t)512 * 64 * 64 * 2);
  _Float16* xc_b   = (_Float16*)alloc((size_t)512 * 64 * 64 * 2);
  _Float16* WihF   = (_Float16*)alloc((size_t)64 * 1024 * 2);
  _Float16* WhhF   = (_Float16*)alloc((size_t)256 * 1024 * 2);
  _Float16* WihB   = (_Float16*)alloc((size_t)64 * 1024 * 2);
  _Float16* WhhB   = (_Float16*)alloc((size_t)256 * 1024 * 2);
  _Float16* hs_f   = (_Float16*)alloc((size_t)512 * 64 * 256 * 2);
  _Float16* hs_b   = (_Float16*)alloc((size_t)512 * 64 * 256 * 2);
  _Float16* subc   = (_Float16*)alloc((size_t)8192 * 512 * 2);
  _Float16* hwW    = (_Float16*)alloc((size_t)512 * 1024 * 2);
  float*    TG     = (float*)   alloc((size_t)8192 * 1024 * 4);
  _Float16* wbuf   = (_Float16*)alloc((size_t)8192 * 640 * 2);
  _Float16* WihWF  = (_Float16*)alloc((size_t)640 * 1024 * 2);
  _Float16* WhhWF  = (_Float16*)alloc((size_t)256 * 1024 * 2);
  _Float16* WihWB  = (_Float16*)alloc((size_t)640 * 1024 * 2);
  _Float16* WhhWB  = (_Float16*)alloc((size_t)256 * 1024 * 2);
  _Float16* whbuf  = (_Float16*)alloc((size_t)8192 * 512 * 2);
  _Float16* emWk   = (_Float16*)alloc((size_t)512 * 64 * 2);
  (void)ws_size; (void)n_in; (void)in_sizes; (void)out_size;

  auto conv = [&](const float* src, _Float16* dst, int Mout, int Kin, int Kpad,
                  int Mpad, int Ntot, int moff) {
    int total = Kpad * Mpad;
    wconvert_swz_kernel<<<(total + 255) / 256, 256, 0, stream>>>(src, dst, Mout, Kin, Kpad,
                                                                 Mpad, Ntot, moff);
  };
  // weight conversions to swizzled fragment-major f16 tiles
  conv(fc_Wih, WihF, 1024, 64, 64, 1024, 1024, 0);
  conv(fc_Whh, WhhF, 1024, 256, 256, 1024, 1024, 0);
  conv(bc_Wih, WihB, 1024, 64, 64, 1024, 1024, 0);
  conv(bc_Whh, WhhB, 1024, 256, 256, 1024, 1024, 0);
  conv(hw_Wt, hwW, 512, 512, 512, 512, 1024, 0);
  conv(hw_Wg, hwW, 512, 512, 512, 512, 1024, 512);
  conv(wf_Wih, WihWF, 1024, 612, 640, 1024, 1024, 0);
  conv(wf_Whh, WhhWF, 1024, 256, 256, 1024, 1024, 0);
  conv(wb_Wih, WihWB, 1024, 612, 640, 1024, 1024, 0);
  conv(wb_Whh, WhhWB, 1024, 256, 256, 1024, 1024, 0);
  conv(em_W, emWk, 50, 512, 512, 64, 64, 0);

  // char embeddings
  charembed_kernel<<<(512 * 64 * 64 + 255) / 256, 256, 0, stream>>>(cmaps_f, ce_W, xc_f);
  charembed_kernel<<<(512 * 64 * 64 + 255) / 256, 256, 0, stream>>>(cmaps_b, ce_W, xc_b);

  // char LSTMs: both dirs + 4 row-shards concurrently (8 WGPs)
  char_lstm_kernel<<<dim3(4, 2), 512, 0, stream>>>(
      xc_f, xc_b, WihF, WhhF, fc_b, cmakers_f, hs_f, WihB, WhhB, bc_b, cmakers_b, hs_b);

  // stable compaction -> subc (B*T, 512)
  compact_kernel<<<dim3(64, 2), 512, 0, stream>>>(hs_f, hs_b, subc);

  // highway GEMM (T and G fused as 1024 output cols)
  hwgemm_kernel<<<2048, 512, 0, stream>>>(subc, hwW, TG);

  // assemble word-BLSTM input buffer
  wordembed_kernel<<<(8192 * 128 + 255) / 256, 256, 0, stream>>>(wmap, we_W, wbuf);
  highway_kernel<<<(8192 * 512 + 255) / 256, 256, 0, stream>>>(TG, subc, hw_bt, hw_bg, wbuf);

  // word BLSTM: both dirs + 8 row-shards concurrently (16 WGPs)
  word_lstm_kernel<<<dim3(8, 2), 512, 0, stream>>>(
      wbuf, WihWF, WhhWF, wf_b, WihWB, WhhWB, wb_b, whbuf);

  // emission + CRF broadcast
  emission_kernel<<<64, 256, 0, stream>>>(whbuf, emWk, em_b, trans, out);

  // tuple tail (ints as floats)
  tail_kernel<<<(16448 + 255) / 256, 256, 0, stream>>>(tmaps, lengths, tmap_corr, out);
}
